// BinLinearBlock_67568425500676
// MI455X (gfx1250) — 8-bit matrix operand order repaired and checked by reading, compiled, not run on hardware
//
#include <hip/hip_runtime.h>
#include <stdint.h>

typedef __attribute__((ext_vector_type(16))) int   v16i;
typedef __attribute__((ext_vector_type(8)))  float v8f;

#define NROWS 65536
#define CIN   1024
#define COUT  1024
#define SLABS 64
#define SLABROWS 1024
constexpr float EPSV = 1e-5f;

#define FP8_P1 0x38u
#define FP8_M1 0xB8u

__global__ void colstats_partial(const float* __restrict__ x,
                                 float* __restrict__ psum,
                                 float* __restrict__ psumsq) {
    int col  = blockIdx.x * blockDim.x + threadIdx.x;
    int slab = blockIdx.y;
    size_t row0 = (size_t)slab * SLABROWS;
    float s = 0.f, ss = 0.f;
    for (int r = 0; r < SLABROWS; ++r) {
        float v = x[(row0 + r) * CIN + col];
        s  += v;
        ss = fmaf(v, v, ss);
    }
    psum  [slab * CIN + col] = s;
    psumsq[slab * CIN + col] = ss;
}

__global__ void colstats_final(const float* __restrict__ psum,
                               const float* __restrict__ psumsq,
                               const float* __restrict__ gamma,
                               const float* __restrict__ beta,
                               float* __restrict__ murs) {
    int col = blockIdx.x * blockDim.x + threadIdx.x;
    float s = 0.f, ss = 0.f;
    for (int i = 0; i < SLABS; ++i) {
        s  += psum  [i * CIN + col];
        ss += psumsq[i * CIN + col];
    }
    const float invN = 1.0f / (float)NROWS;
    float mu  = s * invN;
    float var = fmaf(-mu, mu, ss * invN);
    float rs  = rsqrtf(var + EPSV);
    murs[          col] = mu;
    murs[  CIN +   col] = rs * gamma[col];
    murs[2*CIN +   col] = beta[col];
}

__global__ void pack_x(const float* __restrict__ x,
                       const float* __restrict__ murs,
                       uint8_t* __restrict__ xa) {
    size_t base = ((size_t)blockIdx.x * blockDim.x + threadIdx.x) * 4;
    int col = (int)(base & (CIN - 1));
    float4 v  = *(const float4*)(x + base);
    float4 mu = *(const float4*)(murs + col);
    float4 a  = *(const float4*)(murs + CIN + col);
    float4 bt = *(const float4*)(murs + 2*CIN + col);
    float h0 = fmaf(v.x - mu.x, a.x, bt.x);
    float h1 = fmaf(v.y - mu.y, a.y, bt.y);
    float h2 = fmaf(v.z - mu.z, a.z, bt.z);
    float h3 = fmaf(v.w - mu.w, a.w, bt.w);
    uint32_t o = (h0 >= 0.f ? FP8_P1 : FP8_M1)
               | ((h1 >= 0.f ? FP8_P1 : FP8_M1) << 8)
               | ((h2 >= 0.f ? FP8_P1 : FP8_M1) << 16)
               | ((h3 >= 0.f ? FP8_P1 : FP8_M1) << 24);
    *(uint32_t*)(xa + base) = o;
}

__global__ void pack_w(const float* __restrict__ W, uint8_t* __restrict__ wb) {
    size_t base = ((size_t)blockIdx.x * blockDim.x + threadIdx.x) * 4;
    float4 v = *(const float4*)(W + base);
    uint32_t o = (v.x >= 0.f ? FP8_P1 : FP8_M1)
               | ((v.y >= 0.f ? FP8_P1 : FP8_M1) << 8)
               | ((v.z >= 0.f ? FP8_P1 : FP8_M1) << 16)
               | ((v.w >= 0.f ? FP8_P1 : FP8_M1) << 24);
    *(uint32_t*)(wb + base) = o;
}

__global__ __launch_bounds__(256)
void bgemm(const uint8_t* __restrict__ xa, const uint8_t* __restrict__ wb,
           const float* __restrict__ bias, const float* __restrict__ scale,
           const float* __restrict__ alphap, float* __restrict__ out) {
    __shared__ uint8_t As[2][128 * 128];
    __shared__ uint8_t Bs[2][128 * 128];

    const int tid  = threadIdx.x;
    const int lane = tid & 31;
    const int wave = tid >> 5;
    const int wm   = wave & 1;
    const int wn   = wave >> 1;

    const int row_base = blockIdx.y * 128;
    const int col_base = blockIdx.x * 128;

    const int lrow = lane & 15;
    const int lhi  = lane >> 4;

    const int r0  = tid >> 3;
    const int co  = (tid & 7) * 16;

    v8f acc[4][2];
#pragma unroll
    for (int m = 0; m < 4; ++m)
#pragma unroll
        for (int n = 0; n < 2; ++n)
            acc[m][n] = (v8f){};

    auto stage = [&](int k0, int buf) {
#pragma unroll
        for (int i = 0; i < 4; ++i) {
            int r = r0 + 32 * i;
            uint32_t lA = (uint32_t)(uintptr_t)&As[buf][r * 128 + co];
            uint32_t lB = (uint32_t)(uintptr_t)&Bs[buf][r * 128 + co];
            uint64_t gA = (uint64_t)(uintptr_t)(xa + (size_t)(row_base + r) * CIN + k0 + co);
            uint64_t gB = (uint64_t)(uintptr_t)(wb + (size_t)(col_base + r) * CIN + k0 + co);
            asm volatile("global_load_async_to_lds_b128 %0, %1, off"
                         :: "v"(lA), "v"(gA) : "memory");
            asm volatile("global_load_async_to_lds_b128 %0, %1, off"
                         :: "v"(lB), "v"(gB) : "memory");
        }
    };

    stage(0, 0);
    asm volatile("s_wait_asynccnt 0x0" ::: "memory");
    __syncthreads();

    for (int it = 0; it < CIN / 128; ++it) {
        const int buf = it & 1;
        if (it + 1 < CIN / 128)
            stage((it + 1) * 128, buf ^ 1);

        v16i bf[2];
#pragma unroll
        for (int nf = 0; nf < 2; ++nf) {
            int ocol = wn * 32 + nf * 16 + lrow;
#pragma unroll
            for (int g = 0; g < 4; ++g) {
                int4 d = *(const int4*)(&Bs[buf][ocol * 128 + 32 * g + 16 * lhi]);
                bf[nf][4 * g + 0] = d.x;
                bf[nf][4 * g + 1] = d.y;
                bf[nf][4 * g + 2] = d.z;
                bf[nf][4 * g + 3] = d.w;
            }
        }

#pragma unroll
        for (int mf = 0; mf < 4; ++mf) {
            int arow = wm * 64 + mf * 16 + lrow;
            v16i af;
#pragma unroll
            for (int h = 0; h < 2; ++h)
#pragma unroll
                for (int q = 0; q < 4; ++q) {
                    int2 d = *(const int2*)(&As[buf][arow * 128 + 64 * h + 32 * (q >> 1) + 8 * (q & 1) + 16 * lhi]);
                    af[8 * h + 2 * q + 0] = d.x;
                    af[8 * h + 2 * q + 1] = d.y;
                }
#pragma unroll
            for (int nf = 0; nf < 2; ++nf)
                acc[mf][nf] = __builtin_amdgcn_wmma_f32_16x16x128_fp8_fp8(
                    af, bf[nf], (short)0, acc[mf][nf],
                     false,  false);
        }

        asm volatile("s_wait_asynccnt 0x0" ::: "memory");
        __syncthreads();
    }

    const float alpha = alphap[0];
#pragma unroll
    for (int mf = 0; mf < 4; ++mf) {
#pragma unroll
        for (int nf = 0; nf < 2; ++nf) {
            int n  = col_base + wn * 32 + nf * 16 + lrow;
            float bb = bias[n];
            float sc = scale[n];
            int m0 = row_base + wm * 64 + mf * 16 + 8 * lhi;
#pragma unroll
            for (int r = 0; r < 8; ++r) {
                float y = (acc[mf][nf][r] + bb) * sc;
                y = (y > 0.f) ? y : alpha * y;
                out[(size_t)(m0 + r) * COUT + n] = y;
            }
        }
    }
}

extern "C" void kernel_launch(void* const* d_in, const int* in_sizes, int n_in,
                              void* d_out, int out_size, void* d_ws, size_t ws_size,
                              hipStream_t stream) {
    const float* x     = (const float*)d_in[0];
    const float* gamma = (const float*)d_in[1];
    const float* beta  = (const float*)d_in[2];
    const float* W     = (const float*)d_in[3];
    const float* b     = (const float*)d_in[4];
    const float* scale = (const float*)d_in[5];
    const float* alpha = (const float*)d_in[6];
    float* out = (float*)d_out;

    uint8_t* ws = (uint8_t*)d_ws;
    uint8_t* xa = ws;
    uint8_t* wb = ws + (size_t)NROWS * CIN;
    float* psum   = (float*)(wb + (size_t)COUT * CIN);
    float* psumsq = psum + SLABS * CIN;
    float* murs   = psumsq + SLABS * CIN;

    colstats_partial<<<dim3(CIN / 256, SLABS), 256, 0, stream>>>(x, psum, psumsq);
    colstats_final<<<CIN / 256, 256, 0, stream>>>(psum, psumsq, gamma, beta, murs);
    pack_x<<<((size_t)NROWS * CIN) / (4 * 256), 256, 0, stream>>>(x, murs, xa);
    pack_w<<<((size_t)COUT * CIN) / (4 * 256), 256, 0, stream>>>(W, wb);
    bgemm<<<dim3(COUT / 128, NROWS / 128), 256, 0, stream>>>(xa, wb, b, scale, alpha, out);
}
